// TensorParallelAttention_29858612642593
// MI455X (gfx1250) — matrix operand order repaired and checked by reading, not compiled, not run on hardware
//
#include <hip/hip_runtime.h>
#include <cstdint>

// ---------------------------------------------------------------------------
// TensorParallelAttention for MI455X (gfx1250, wave32)
//   k0: fp32 -> bf16 conversion (X, W_qkv, W_out)      [one pass, HBM-bound]
//   k1: QKV GEMM, 32x64 per wave, bf16 WMMA            [matrix-pipe bound]
//   k2: fused causal flash attention; K/V tiles double-buffered in LDS via
//       the Tensor Data Mover (tensor_load_to_lds, s_wait_tensorcnt pipeline)
//   k3: output projection GEMM, 32x64 per wave
// All matmuls: V_WMMA_F32_16X16X32_BF16 (f32 accumulate).
// ---------------------------------------------------------------------------

constexpr int Hh   = 1024;   // hidden
constexpr int NHh  = 16;     // heads
constexpr int HDd  = 64;     // head dim
constexpr int SEQq = 2048;
constexpr int Bb   = 2;
constexpr int TOK  = Bb * SEQq;  // 4096 tokens

typedef unsigned int u32;
typedef unsigned long long u64;
typedef __attribute__((ext_vector_type(16))) __bf16 v16bf;
typedef __attribute__((ext_vector_type(8)))  __bf16 bf16x8;
typedef __attribute__((ext_vector_type(8)))  float  v8f;
typedef __attribute__((ext_vector_type(4)))  float  f32x4;
typedef __attribute__((ext_vector_type(4)))  u32    u32x4;
typedef __attribute__((ext_vector_type(8)))  u32    u32x8;

static __device__ __forceinline__ int lane_id() { return (int)(threadIdx.x & 31u); }

static __device__ __forceinline__ v8f wmma_bf16(v16bf a, v16bf b, v8f c) {
  return __builtin_amdgcn_wmma_f32_16x16x32_bf16(false, a, false, b, (short)0, c,
                                                 false, false);
}

// A fragment (16x32 MxK bf16) from row-major bf16, row stride ld elements.
static __device__ __forceinline__ v16bf load_a_bf16(const __bf16* base, int ld,
                                                    int row0, int k0) {
  int l = lane_id(); int hf = l >> 4; int lr = l & 15;
  const __bf16* p = base + (size_t)(row0 + lr) * ld + k0 + hf * 8;
  bf16x8 lo = *reinterpret_cast<const bf16x8*>(p);
  bf16x8 hi = *reinterpret_cast<const bf16x8*>(p + 16);
  v16bf a;
#pragma unroll
  for (int i = 0; i < 8; ++i) { a[i] = lo[i]; a[8 + i] = hi[i]; }
  return a;
}

// B fragment (32x16 KxN bf16): B[k][n] = BT[n][k], BT row-major, stride ld.
static __device__ __forceinline__ v16bf load_b_bf16(const __bf16* bt, int ld,
                                                    int n0, int k0) {
  int l = lane_id();
  const __bf16* p = bt + (size_t)(n0 + (l & 15)) * ld + k0 + (l >> 4) * 8;
  bf16x8 lo = *reinterpret_cast<const bf16x8*>(p);
  bf16x8 hi = *reinterpret_cast<const bf16x8*>(p + 16);
  v16bf b;
#pragma unroll
  for (int i = 0; i < 8; ++i) { b[i] = lo[i]; b[8 + i] = hi[i]; }
  return b;
}

// ---------------------------------------------------------------------------
// Tensor Data Mover: 2D tile load, bf16 elements (data_size code 1 = 2 bytes).
// D# per CDNA5 ISA. Groups 2/3 zero (2D tensor).
// ---------------------------------------------------------------------------
static __device__ __forceinline__ void tdm_load_2d_bf16(const void* gptr, u32 lds_off,
                                                        u32 tensor_d0, u32 tensor_d1,
                                                        u32 tile_d0, u32 tile_d1,
                                                        u64 stride0) {
  u64 ga = (u64)(uintptr_t)gptr;
  u32x4 g0;
  g0[0] = 1u;                                           // count=1, user descriptor
  g0[1] = lds_off;                                      // LDS byte address
  g0[2] = (u32)ga;                                      // global addr [31:0]
  g0[3] = (u32)((ga >> 32) & 0x1FFFFFFu) | (2u << 30);  // addr[56:32] | type=2
  u32x8 g1;
  g1[0] = 1u << 16;                                     // data_size=1 (2B); mask=0
  g1[1] = (tensor_d0 & 0xFFFFu) << 16;                  // tensor_dim0[15:0]
  g1[2] = (tensor_d0 >> 16) | ((tensor_d1 & 0xFFFFu) << 16);
  g1[3] = (tensor_d1 >> 16) | (tile_d0 << 16);          // dim1 hi | tile_dim0
  g1[4] = tile_d1;                                      // tile_dim1; tile_dim2=0
  g1[5] = (u32)stride0;                                 // dim0 stride [31:0]
  g1[6] = (u32)((stride0 >> 32) & 0xFFFFu);             // dim0 stride [47:32]
  g1[7] = 0u;                                           // dim1 stride unused (2D)
  u32x4 gz = {0u, 0u, 0u, 0u};
  asm volatile("tensor_load_to_lds %0, %1, %2, %3"
               :: "s"(g0), "s"(g1), "s"(gz), "s"(gz)
               : "memory");
}

// ---------------------------------------------------------------------------
// Kernel 0: fp32 -> bf16 bulk convert (8 elements / thread).
// ---------------------------------------------------------------------------
__global__ __launch_bounds__(256)
void cvt_f32_bf16_kernel(const float* __restrict__ in, __bf16* __restrict__ out,
                         int n8) {
  int i = blockIdx.x * 256 + threadIdx.x;
  if (i >= n8) return;
  const f32x4* p = reinterpret_cast<const f32x4*>(in) + (size_t)i * 2;
  f32x4 x0 = p[0], x1 = p[1];
  bf16x8 o;
#pragma unroll
  for (int k = 0; k < 4; ++k) { o[k] = (__bf16)x0[k]; o[4 + k] = (__bf16)x1[k]; }
  reinterpret_cast<bf16x8*>(out)[i] = o;
}

// ---------------------------------------------------------------------------
// Kernel 1: QKV projection, 32x64 tile per wave (8 WMMA / K-step).
// Scatter: n -> head=n/192, r=n%192: r<64 Q[b,h,s,d]; r<128 K; else Vt[b,h,d,s].
// ---------------------------------------------------------------------------
__global__ __launch_bounds__(256)
void qkv_gemm_kernel(const __bf16* __restrict__ Xb, const __bf16* __restrict__ Wb,
                     const float* __restrict__ bqkv, __bf16* __restrict__ Qw,
                     __bf16* __restrict__ Kw, __bf16* __restrict__ Vtw) {
  const int NT = (3 * Hh) / 64;  // 48 column tiles of 64
  int tile = blockIdx.x * 8 + (threadIdx.x >> 5);
  int mt = tile / NT, nt = tile % NT;
  int m0 = mt * 32, n0 = nt * 64;

  v8f c[2][4];
#pragma unroll
  for (int i = 0; i < 2; ++i)
#pragma unroll
    for (int j = 0; j < 4; ++j) c[i][j] = (v8f){0.f,0.f,0.f,0.f,0.f,0.f,0.f,0.f};

  for (int k0 = 0; k0 < Hh; k0 += 32) {
    v16bf a0 = load_a_bf16(Xb, Hh, m0, k0);
    v16bf a1 = load_a_bf16(Xb, Hh, m0 + 16, k0);
#pragma unroll
    for (int j = 0; j < 4; ++j) {
      v16bf b = load_b_bf16(Wb, Hh, n0 + j * 16, k0);
      c[0][j] = wmma_bf16(a0, b, c[0][j]);
      c[1][j] = wmma_bf16(a1, b, c[1][j]);
    }
  }

  int l = lane_id(); int hf = l >> 4; int ln = l & 15;
#pragma unroll
  for (int j = 0; j < 4; ++j) {
    int n    = n0 + j * 16 + ln;
    float bias = bqkv[n];
    int head = n / (3 * HDd);
    int r    = n % (3 * HDd);
    int type = r >> 6;          // 0=q 1=k 2=v
    int d    = r & (HDd - 1);
#pragma unroll
    for (int i = 0; i < 2; ++i) {
#pragma unroll
      for (int vr = 0; vr < 8; ++vr) {
        int m  = m0 + i * 16 + vr + 8 * hf;   // token index
        int bb = m / SEQq;
        int s  = m % SEQq;
        size_t bh = (size_t)(bb * NHh + head);
        __bf16 bv = (__bf16)(c[i][j][vr] + bias);
        if (type == 0)      Qw[(bh * SEQq + s) * HDd + d]  = bv;
        else if (type == 1) Kw[(bh * SEQq + s) * HDd + d]  = bv;
        else                Vtw[(bh * HDd + d) * SEQq + s] = bv;
      }
    }
  }
}

// ---------------------------------------------------------------------------
// Kernel 2: fused causal flash attention per (b, head, 64 queries).
// 4 waves/block, each owns 16 queries; uniform 64-key chunk loop across the
// block. K (64x64) and V^T (64x64) tiles double-buffered in LDS and streamed
// by the TDM: chunk kc+1 is issued before computing chunk kc, and
// s_wait_tensorcnt(2) retires only the current chunk's two in-order loads.
// ---------------------------------------------------------------------------
__global__ __launch_bounds__(128)
void attn_kernel(const __bf16* __restrict__ Qw, const __bf16* __restrict__ Kw,
                 const __bf16* __restrict__ Vtw, __bf16* __restrict__ attnOut) {
  __shared__ __bf16 Klds[2][64 * 64];   // [buf][key][d]
  __shared__ __bf16 Vlds[2][64 * 64];   // [buf][d][key]
  __shared__ __bf16 Plds[4][16 * 64];   // per-wave P tile

  const int wave  = threadIdx.x >> 5;
  const int l     = lane_id();
  const int hf    = l >> 4;
  const int ln    = l & 15;
  const int qb64  = blockIdx.x % (SEQq / 64);
  const int head  = (blockIdx.x / (SEQq / 64)) % NHh;
  const int batch = blockIdx.x / ((SEQq / 64) * NHh);
  const int qbase = qb64 * 64 + wave * 16;

  const __bf16* Qb = Qw  + (size_t)(batch * NHh + head) * SEQq * HDd;
  const __bf16* Kb = Kw  + (size_t)(batch * NHh + head) * SEQq * HDd;
  const __bf16* Vb = Vtw + (size_t)(batch * NHh + head) * HDd * SEQq;

  const u32 ldsK0 = (u32)(uintptr_t)&Klds[0][0];
  const u32 ldsK1 = (u32)(uintptr_t)&Klds[1][0];
  const u32 ldsV0 = (u32)(uintptr_t)&Vlds[0][0];
  const u32 ldsV1 = (u32)(uintptr_t)&Vlds[1][0];

  // Q tile resident in registers (16x64 as two 16x32 A-fragments).
  v16bf qa0 = load_a_bf16(Qb, HDd, qbase, 0);
  v16bf qa1 = load_a_bf16(Qb, HDd, qbase, 32);

  v8f o[4];
#pragma unroll
  for (int j = 0; j < 4; ++j) o[j] = (v8f){0.f,0.f,0.f,0.f,0.f,0.f,0.f,0.f};
  float mrun[8], lrun[8];
#pragma unroll
  for (int r = 0; r < 8; ++r) { mrun[r] = -1e30f; lrun[r] = 0.f; }

  const int nch = qb64 + 1;  // uniform across block; causal upper bound

  // prologue: prefetch chunk 0 into buffer 0
  if (wave == 0) {
    tdm_load_2d_bf16(Kb, ldsK0, HDd, SEQq, 64, 64, HDd);
    tdm_load_2d_bf16(Vb, ldsV0, SEQq, HDd, 64, 64, SEQq);
  }

  for (int kc = 0; kc < nch; ++kc) {
    const int kbase = kc * 64;
    const int buf = kc & 1;
    const __bf16* Kt = &Klds[buf][0];
    const __bf16* Vt = &Vlds[buf][0];

    if (wave == 0) {
      if (kc + 1 < nch) {
        // prefetch next chunk into the other buffer (freed by last barrier)
        const int nb = kbase + 64;
        tdm_load_2d_bf16(Kb + (size_t)nb * HDd, buf ? ldsK0 : ldsK1,
                         HDd, SEQq, 64, 64, HDd);
        tdm_load_2d_bf16(Vb + nb, buf ? ldsV0 : ldsV1,
                         SEQq, HDd, 64, 64, SEQq);
        __builtin_amdgcn_s_wait_tensorcnt(2);  // current chunk landed
      } else {
        __builtin_amdgcn_s_wait_tensorcnt(0);
      }
    }
    __syncthreads();

    // scores S = Q @ K^T : four 16x16 key tiles, K-dim = 64
    v8f s[4];
#pragma unroll
    for (int t = 0; t < 4; ++t) {
      s[t] = (v8f){0.f,0.f,0.f,0.f,0.f,0.f,0.f,0.f};
      s[t] = wmma_bf16(qa0, load_b_bf16(Kt, 64, t * 16, 0),  s[t]);
      s[t] = wmma_bf16(qa1, load_b_bf16(Kt, 64, t * 16, 32), s[t]);
    }

    // scale 1/sqrt(64) + causal mask
#pragma unroll
    for (int t = 0; t < 4; ++t) {
      int key = kbase + t * 16 + ln;
#pragma unroll
      for (int r = 0; r < 8; ++r) {
        int q = qbase + r + 8 * hf;
        float v = s[t][r] * 0.125f;
        s[t][r] = (key > q) ? -1e30f : v;
      }
    }

    // online softmax (row r+8*hf lives across a 16-lane half)
    float alpha[8];
#pragma unroll
    for (int r = 0; r < 8; ++r) {
      float mx = fmaxf(fmaxf(s[0][r], s[1][r]), fmaxf(s[2][r], s[3][r]));
      mx = fmaxf(mx, __shfl_xor(mx, 1, 32));
      mx = fmaxf(mx, __shfl_xor(mx, 2, 32));
      mx = fmaxf(mx, __shfl_xor(mx, 4, 32));
      mx = fmaxf(mx, __shfl_xor(mx, 8, 32));
      float newm = fmaxf(mrun[r], mx);
      float al = __expf(mrun[r] - newm);
      float p0 = __expf(s[0][r] - newm);
      float p1 = __expf(s[1][r] - newm);
      float p2 = __expf(s[2][r] - newm);
      float p3 = __expf(s[3][r] - newm);
      float rs = (p0 + p1) + (p2 + p3);
      rs += __shfl_xor(rs, 1, 32);
      rs += __shfl_xor(rs, 2, 32);
      rs += __shfl_xor(rs, 4, 32);
      rs += __shfl_xor(rs, 8, 32);
      lrun[r] = lrun[r] * al + rs;
      mrun[r] = newm;
      alpha[r] = al;
      s[0][r] = p0; s[1][r] = p1; s[2][r] = p2; s[3][r] = p3;
    }
#pragma unroll
    for (int j = 0; j < 4; ++j)
#pragma unroll
      for (int r = 0; r < 8; ++r) o[j][r] *= alpha[r];

    // P (C layout) -> LDS row-major bf16 [16 x 64]
    __bf16* Pw = &Plds[wave][0];
#pragma unroll
    for (int t = 0; t < 4; ++t)
#pragma unroll
      for (int r = 0; r < 8; ++r) {
        int m = r + 8 * hf;
        Pw[m * 64 + t * 16 + ln] = (__bf16)s[t][r];
      }
    asm volatile("s_wait_dscnt 0" ::: "memory");

    // P back as two 16x32 A-fragments
    v16bf pa0 = load_a_bf16(Pw, 64, 0, 0);
    v16bf pa1 = load_a_bf16(Pw, 64, 0, 32);

    // O += P @ V  (V^T tile in LDS -> contiguous B-fragments, K-dim = keys)
#pragma unroll
    for (int j = 0; j < 4; ++j) {
      o[j] = wmma_bf16(pa0, load_b_bf16(Vt, 64, j * 16, 0),  o[j]);
      o[j] = wmma_bf16(pa1, load_b_bf16(Vt, 64, j * 16, 32), o[j]);
    }
    __syncthreads();  // all waves done with this buffer before reuse
  }

  // epilogue: normalize and store attn as bf16 in [B,S,H] token-major layout
  __bf16* outb = attnOut + (size_t)batch * SEQq * Hh;
#pragma unroll
  for (int r = 0; r < 8; ++r) {
    int m = qbase + r + 8 * hf;
    float inv = 1.0f / lrun[r];
    __bf16* row = outb + (size_t)m * Hh + head * HDd;
#pragma unroll
    for (int j = 0; j < 4; ++j) row[j * 16 + ln] = (__bf16)(o[j][r] * inv);
  }
}

// ---------------------------------------------------------------------------
// Kernel 3: output projection, 32x64 tile per wave, fp32 result.
// ---------------------------------------------------------------------------
__global__ __launch_bounds__(256)
void out_gemm_kernel(const __bf16* __restrict__ Ab, const __bf16* __restrict__ Wb,
                     float* __restrict__ out) {
  const int NT = Hh / 64;  // 16 column tiles
  int tile = blockIdx.x * 8 + (threadIdx.x >> 5);
  int mt = tile / NT, nt = tile % NT;
  int m0 = mt * 32, n0 = nt * 64;

  v8f c[2][4];
#pragma unroll
  for (int i = 0; i < 2; ++i)
#pragma unroll
    for (int j = 0; j < 4; ++j) c[i][j] = (v8f){0.f,0.f,0.f,0.f,0.f,0.f,0.f,0.f};

  for (int k0 = 0; k0 < Hh; k0 += 32) {
    v16bf a0 = load_a_bf16(Ab, Hh, m0, k0);
    v16bf a1 = load_a_bf16(Ab, Hh, m0 + 16, k0);
#pragma unroll
    for (int j = 0; j < 4; ++j) {
      v16bf b = load_b_bf16(Wb, Hh, n0 + j * 16, k0);
      c[0][j] = wmma_bf16(a0, b, c[0][j]);
      c[1][j] = wmma_bf16(a1, b, c[1][j]);
    }
  }

  int l = lane_id(); int hf = l >> 4; int ln = l & 15;
#pragma unroll
  for (int i = 0; i < 2; ++i)
#pragma unroll
    for (int j = 0; j < 4; ++j)
#pragma unroll
      for (int vr = 0; vr < 8; ++vr) {
        int m = m0 + i * 16 + vr + 8 * hf;
        out[(size_t)m * Hh + n0 + j * 16 + ln] = c[i][j][vr];
      }
}

// ---------------------------------------------------------------------------
extern "C" void kernel_launch(void* const* d_in, const int* in_sizes, int n_in,
                              void* d_out, int out_size, void* d_ws, size_t ws_size,
                              hipStream_t stream) {
  (void)in_sizes; (void)n_in; (void)out_size; (void)ws_size;

  const float* X    = (const float*)d_in[0];  // hidden_states [B,S,H]
  //             d_in[1] = attention_mask (causal; implemented analytically)
  const float* Wqkv = (const float*)d_in[2];  // [3H, H]
  const float* bqkv = (const float*)d_in[3];  // [3H]
  const float* Wout = (const float*)d_in[4];  // [H, H]
  float* out = (float*)d_out;                 // [B,S,H]

  // Workspace (all bf16): Xb | Wqkvb | Woutb | Q | K | V^T | attn
  const size_t nX   = (size_t)TOK * Hh;        // 4M
  const size_t nWq  = (size_t)3 * Hh * Hh;     // 3M
  const size_t nWo  = (size_t)Hh * Hh;         // 1M
  const size_t nQKV = (size_t)Bb * NHh * SEQq * HDd;  // 4M each
  __bf16* Xb    = (__bf16*)d_ws;
  __bf16* Wqkvb = Xb + nX;
  __bf16* Woutb = Wqkvb + nWq;
  __bf16* Qw    = Woutb + nWo;
  __bf16* Kw    = Qw + nQKV;
  __bf16* Vtw   = Kw + nQKV;
  __bf16* attnB = Vtw + nQKV;

  // k0: bulk fp32 -> bf16
  cvt_f32_bf16_kernel<<<(int)(nX  / 8 / 256), 256, 0, stream>>>(X, Xb, (int)(nX / 8));
  cvt_f32_bf16_kernel<<<(int)(nWq / 8 / 256), 256, 0, stream>>>(Wqkv, Wqkvb, (int)(nWq / 8));
  cvt_f32_bf16_kernel<<<(int)(nWo / 8 / 256), 256, 0, stream>>>(Wout, Woutb, (int)(nWo / 8));

  // k1: QKV projection
  {
    int tiles = (TOK / 32) * ((3 * Hh) / 64);  // 6144 wave tiles
    qkv_gemm_kernel<<<tiles / 8, 256, 0, stream>>>(Xb, Wqkvb, bqkv, Qw, Kw, Vtw);
  }
  // k2: fused attention
  {
    int blocks = Bb * NHh * (SEQq / 64);       // 1024
    attn_kernel<<<blocks, 128, 0, stream>>>(Qw, Kw, Vtw, attnB);
  }
  // k3: output projection
  {
    int tiles = (TOK / 32) * (Hh / 64);        // 2048 wave tiles
    out_gemm_kernel<<<tiles / 8, 256, 0, stream>>>(attnB, Woutb, out);
  }
}
